// TritonHSTUAttention_67680094650972
// MI455X (gfx1250) — matrix operand order repaired and checked by reading, not compiled, not run on hardware
//
#include <hip/hip_runtime.h>
#include <hip/hip_bf16.h>
#include <stdint.h>

// HSTU jagged causal attention, fp32 I/O, bf16 WMMA compute (gfx1250).
#define NHEAD 8
#define DQK   128
#define DVV   128
#define MTILE 128   // query rows per block (8 waves x 16 rows)
#define JTILE 32    // key rows per j-step (matches WMMA K=32 for 2nd GEMM)
#define MAXSEQ 1024

typedef __attribute__((ext_vector_type(16))) __bf16 v16bf;
typedef __attribute__((ext_vector_type(8)))  float  v8f;

union FragB {
  v16bf    v;      // 16 bf16 = 32 bytes = 8 VGPRs
  uint32_t u[8];
  uint4    q[2];
};

// pack two fp32 -> bf16x2 (RNE). Prefer native conversion hardware.
__device__ __forceinline__ uint32_t pkbf(float lo, float hi) {
#if __has_builtin(__builtin_amdgcn_cvt_pk_bf16_f32)
  typedef __attribute__((ext_vector_type(2))) __bf16 v2bf;
  union { v2bf v; uint32_t u; } cv;
  cv.v = __builtin_amdgcn_cvt_pk_bf16_f32(lo, hi);
  return cv.u;
#else
  // fptrunc f32->bf16: native on gfx1250 (v_cvt_pk_bf16_f32 / v_cvt_bf16_f32)
  union { __bf16 h[2]; uint32_t u; } cv;
  cv.h[0] = static_cast<__bf16>(lo);
  cv.h[1] = static_cast<__bf16>(hi);
  return cv.u;
#endif
}

// fast silu: x * rcp(1 + exp(-x))  (v_exp_f32 + v_add + v_rcp_f32 + v_mul)
__device__ __forceinline__ float fsilu(float x) {
  return x * __builtin_amdgcn_rcpf(1.0f + __expf(-x));
}

__global__ __launch_bounds__(256)
void hstu_attn_wmma(const float* __restrict__ tq,
                    const float* __restrict__ tk,
                    const float* __restrict__ tv,
                    const int*   __restrict__ offsets,
                    const int*   __restrict__ p_maxseq,
                    const int*   __restrict__ p_scale,
                    float*       __restrict__ out) {
  __shared__ uint16_t Kt[JTILE][DQK];      // bf16, row-major  (8 KB)
  __shared__ uint16_t Vt[DVV][JTILE];      // bf16, transposed (8 KB)
  __shared__ float    Sscr[8][16][32];     // per-wave S transpose scratch (16 KB)

  const int b    = blockIdx.z;
  const int h    = blockIdx.y;
  const int off  = offsets[b];
  const int nlen = offsets[b + 1] - off;
  const int row0 = blockIdx.x * MTILE;
  if (row0 >= nlen) return;                // uniform per block, before barriers

  const float alpha    = 0.08838834764831845f;  // 1/sqrt(128), folded into Q
  const int   ms       = p_maxseq[0];
  const int   sc       = p_scale[0];
  const float invDenom = 1.0f / (float)(sc > 0 ? sc : ms);  // folded into epilogue

  const int tid  = threadIdx.x;
  const int wave = tid >> 5;
  const int lane = tid & 31;
  const int l16  = lane & 15;
  const int hi   = lane >> 4;
  const int qb   = row0 + wave * 16;

  // ---- Q A-fragments (16x32 bf16 each), 4 chunks covering D=128; pre-scaled by alpha
  FragB qf[4];
  {
    int qrow = qb + l16; if (qrow >= nlen) qrow = nlen - 1;
    const float* qp = tq + (size_t)(off + qrow) * (NHEAD * DQK) + h * DQK;
#pragma unroll
    for (int c = 0; c < 4; ++c) {
      const int c0 = c * 32 + hi * 8;     // lanes 0-15: K 0-7/16-23; 16-31: K 8-15/24-31
      float4 x0 = *(const float4*)(qp + c0);
      float4 x1 = *(const float4*)(qp + c0 + 4);
      float4 y0 = *(const float4*)(qp + c0 + 16);
      float4 y1 = *(const float4*)(qp + c0 + 20);
      qf[c].u[0] = pkbf(x0.x * alpha, x0.y * alpha);
      qf[c].u[1] = pkbf(x0.z * alpha, x0.w * alpha);
      qf[c].u[2] = pkbf(x1.x * alpha, x1.y * alpha);
      qf[c].u[3] = pkbf(x1.z * alpha, x1.w * alpha);
      qf[c].u[4] = pkbf(y0.x * alpha, y0.y * alpha);
      qf[c].u[5] = pkbf(y0.z * alpha, y0.w * alpha);
      qf[c].u[6] = pkbf(y1.x * alpha, y1.y * alpha);
      qf[c].u[7] = pkbf(y1.z * alpha, y1.w * alpha);
    }
  }

  v8f acc[8];
#pragma unroll
  for (int i = 0; i < 8; ++i) acc[i] = (v8f){};

  const int jmax = (nlen < row0 + MTILE) ? nlen : (row0 + MTILE);

  for (int j = 0; j < jmax; j += JTILE) {
    // ---- cooperative staging: K row-major bf16 ----
    {
      const int r  = tid >> 3;              // 0..31
      const int cg = (tid & 7) * 16;        // 16-col group
      int krow = j + r; if (krow >= nlen) krow = nlen - 1;
      const float* kp = tk + (size_t)(off + krow) * (NHEAD * DQK) + h * DQK + cg;
      float4 a0 = *(const float4*)(kp + 0);
      float4 a1 = *(const float4*)(kp + 4);
      float4 a2 = *(const float4*)(kp + 8);
      float4 a3 = *(const float4*)(kp + 12);
      if (j + JTILE < jmax)                 // prefetch next tile into near caches
        __builtin_prefetch(kp + (size_t)JTILE * NHEAD * DQK, 0, 3);
      uint32_t* dst = (uint32_t*)&Kt[r][cg];
      dst[0] = pkbf(a0.x, a0.y); dst[1] = pkbf(a0.z, a0.w);
      dst[2] = pkbf(a1.x, a1.y); dst[3] = pkbf(a1.z, a1.w);
      dst[4] = pkbf(a2.x, a2.y); dst[5] = pkbf(a2.z, a2.w);
      dst[6] = pkbf(a3.x, a3.y); dst[7] = pkbf(a3.z, a3.w);
    }
    // ---- cooperative staging: V transposed bf16 (Vt[d][key]) ----
    {
      const int rp = tid >> 4;              // row pair 0..15 -> rows 2rp,2rp+1
      const int cg = (tid & 15) * 8;        // 8 d-cols
      int r0g = j + rp * 2, r1g = r0g + 1;
      if (r0g >= nlen) r0g = nlen - 1;
      if (r1g >= nlen) r1g = nlen - 1;
      const float* v0 = tv + (size_t)(off + r0g) * (NHEAD * DVV) + h * DVV + cg;
      const float* v1 = tv + (size_t)(off + r1g) * (NHEAD * DVV) + h * DVV + cg;
      float4 a0 = *(const float4*)(v0);
      float4 a1 = *(const float4*)(v0 + 4);
      float4 b0 = *(const float4*)(v1);
      float4 b1 = *(const float4*)(v1 + 4);
      if (j + JTILE < jmax)
        __builtin_prefetch(v0 + (size_t)JTILE * NHEAD * DVV, 0, 3);
      float aa[8] = {a0.x, a0.y, a0.z, a0.w, a1.x, a1.y, a1.z, a1.w};
      float bb[8] = {b0.x, b0.y, b0.z, b0.w, b1.x, b1.y, b1.z, b1.w};
#pragma unroll
      for (int c = 0; c < 8; ++c)
        *(uint32_t*)&Vt[cg + c][rp * 2] = pkbf(aa[c], bb[c]);
    }
    __syncthreads();

    // ---- S = Q * K^T for two 16-key subtiles (8 WMMAs) ----
    v8f s0 = (v8f){}, s1 = (v8f){};
#pragma unroll
    for (int c = 0; c < 4; ++c) {
      FragB kf0, kf1;
      const uint4* p0 = (const uint4*)&Kt[l16][c * 32 + hi * 8];
      kf0.q[0] = p0[0]; kf0.q[1] = p0[2];
      const uint4* p1 = (const uint4*)&Kt[16 + l16][c * 32 + hi * 8];
      kf1.q[0] = p1[0]; kf1.q[1] = p1[2];
      s0 = __builtin_amdgcn_wmma_f32_16x16x32_bf16(false, qf[c].v, false, kf0.v,
                                                   (short)0, s0, false, false);
      s1 = __builtin_amdgcn_wmma_f32_16x16x32_bf16(false, qf[c].v, false, kf1.v,
                                                   (short)0, s1, false, false);
    }

    // ---- silu + (mask only on edge tiles); C-layout -> LDS row-major ----
    {
      float* sp = &Sscr[wave][0][0];
      const bool fullTile = (j + JTILE - 1 <= qb) && (j + JTILE <= nlen) &&
                            (qb + 15 < nlen);            // wave-uniform branch
      if (fullTile) {
#pragma unroll
        for (int r = 0; r < 8; ++r) {
          const int m = r + hi * 8;
          sp[m * 32 + l16]      = fsilu(s0[r]);
          sp[m * 32 + 16 + l16] = fsilu(s1[r]);
        }
      } else {
        const int key0 = j + l16;
        const int key1 = key0 + 16;
#pragma unroll
        for (int r = 0; r < 8; ++r) {
          const int m  = r + hi * 8;        // C layout: M = r + 8*(lane>=16)
          const int ig = qb + m;
          float a0v = fsilu(s0[r]);
          float a1v = fsilu(s1[r]);
          bool ok0 = (key0 <= ig) && (key0 < nlen) && (ig < nlen);
          bool ok1 = (key1 <= ig) && (key1 < nlen) && (ig < nlen);
          sp[m * 32 + l16]      = ok0 ? a0v : 0.0f;
          sp[m * 32 + 16 + l16] = ok1 ? a1v : 0.0f;
        }
      }
    }

    // ---- read S back in A-fragment layout (16x32 bf16) ----
    FragB sf;
    {
      const int c0 = hi * 8;
      const float* row = &Sscr[wave][l16][0];
#pragma unroll
      for (int v = 0; v < 4; ++v) {
        float2 p = *(const float2*)(row + c0 + 2 * v);
        sf.u[v] = pkbf(p.x, p.y);
      }
#pragma unroll
      for (int v = 0; v < 4; ++v) {
        float2 p = *(const float2*)(row + c0 + 16 + 2 * v);
        sf.u[4 + v] = pkbf(p.x, p.y);
      }
    }

    // ---- O += S * V (8 WMMAs over DV=128) ----
#pragma unroll
    for (int dc = 0; dc < 8; ++dc) {
      FragB vf;
      const uint4* pv = (const uint4*)&Vt[dc * 16 + l16][hi * 8];
      vf.q[0] = pv[0]; vf.q[1] = pv[2];
      acc[dc] = __builtin_amdgcn_wmma_f32_16x16x32_bf16(false, sf.v, false, vf.v,
                                                        (short)0, acc[dc], false, false);
    }
    __syncthreads();   // protect Kt/Vt before next staging round
  }

  // ---- epilogue: scale by 1/denom, scatter fp32 to jagged output ----
#pragma unroll
  for (int r = 0; r < 8; ++r) {
    const int m  = r + hi * 8;
    const int ig = qb + m;
    if (ig < nlen) {
      float* op = out + (size_t)(off + ig) * (NHEAD * DVV) + h * DVV;
#pragma unroll
      for (int dc = 0; dc < 8; ++dc)
        op[dc * 16 + l16] = acc[dc][r] * invDenom;
    }
  }
}

extern "C" void kernel_launch(void* const* d_in, const int* in_sizes, int n_in,
                              void* d_out, int out_size, void* d_ws, size_t ws_size,
                              hipStream_t stream) {
  const float* tq      = (const float*)d_in[0];
  const float* tk      = (const float*)d_in[1];
  const float* tv      = (const float*)d_in[2];
  const int*   offsets = (const int*)d_in[3];
  const int*   maxseq  = (const int*)d_in[4];
  const int*   scale   = (const int*)d_in[5];
  float*       out     = (float*)d_out;

  const int B = in_sizes[3] - 1;                 // offsets has B+1 entries
  dim3 grid((MAXSEQ + MTILE - 1) / MTILE, NHEAD, B);
  dim3 block(256, 1, 1);
  hstu_attn_wmma<<<grid, block, 0, stream>>>(tq, tk, tv, offsets, maxseq, scale, out);
}
